// BlockLocalAttentionProduct_16965120819263
// MI455X (gfx1250) — matrix operand order repaired and checked by reading, not compiled, not run on hardware
//
#include <hip/hip_runtime.h>

// Block-local attention for MI455X (gfx1250, wave32, WMMA bf16).
// B=4 H=12 T=4096 D=64, chunk CS=256, half-window W=128.
// Kernel A: full-pass attention over 16 chunks, writes d_out.
// Kernel B: shifted-pass over 15 chunks, blends 0.5*old + 0.5*new in place.
// Softmax row-max via DPP row_ror rotate-reduce (VALU, no LDS),
// softmax row-sum via an extra WMMA against an all-ones B tile.

#define NB 4
#define NH 12
#define NT 4096
#define ND 64
#define CS 256
#define HW 128

typedef __attribute__((ext_vector_type(16))) __bf16 v16bf;
typedef __attribute__((ext_vector_type(8)))  float  v8f;
typedef unsigned long long ull;

__device__ __forceinline__ unsigned short f2bf(float f) {
  unsigned int u = __float_as_uint(f);
  u += 0x7FFFu + ((u >> 16) & 1u);          // round-to-nearest-even
  return (unsigned short)(u >> 16);
}

union FragU { v16bf v; unsigned short u[16]; };

// A-operand layout (16-bit, 16xK): lane = row (mod 16);
// elems 0..7 -> k = sel*8 + e ; elems 8..15 -> k = 16 + sel*8 + (e-8)
__device__ __forceinline__ v16bf load_frag_a(const unsigned short* base, int row,
                                             int stride, int k0, int sel) {
  FragU f;
  const unsigned short* p = base + row * stride + k0;
#pragma unroll
  for (int e = 0; e < 8; ++e) f.u[e] = p[sel * 8 + e];
#pragma unroll
  for (int e = 0; e < 8; ++e) f.u[8 + e] = p[16 + sel * 8 + e];
  return f.v;
}

// B-operand layout (16-bit, 32x16): lane = column (mod 16);
// elems 0..7 -> k = sel*8 + e ; elems 8..15 -> k = 16 + sel*8 + (e-8)
__device__ __forceinline__ v16bf load_frag_b(const unsigned short* base, int col,
                                             int stride, int k0, int sel) {
  FragU f;
  const unsigned short* p = base + col * stride + k0;
#pragma unroll
  for (int e = 0; e < 8; ++e) f.u[e] = p[sel * 8 + e];
#pragma unroll
  for (int e = 0; e < 8; ++e) f.u[8 + e] = p[16 + sel * 8 + e];
  return f.v;
}

// Max across each 16-lane DPP row via rotate-reduce (row_ror 1,2,4,8).
// Pure VALU: no LDS-bpermute traffic, no dscnt waits.
__device__ __forceinline__ float rowmax16(float x) {
  int t;
  t = __builtin_amdgcn_update_dpp(0, __float_as_int(x), 0x121, 0xf, 0xf, false); // row_ror:1
  x = fmaxf(x, __int_as_float(t));
  t = __builtin_amdgcn_update_dpp(0, __float_as_int(x), 0x122, 0xf, 0xf, false); // row_ror:2
  x = fmaxf(x, __int_as_float(t));
  t = __builtin_amdgcn_update_dpp(0, __float_as_int(x), 0x124, 0xf, 0xf, false); // row_ror:4
  x = fmaxf(x, __int_as_float(t));
  t = __builtin_amdgcn_update_dpp(0, __float_as_int(x), 0x128, 0xf, 0xf, false); // row_ror:8
  x = fmaxf(x, __int_as_float(t));
  return x;
}

__global__ void __launch_bounds__(512)
blk_attn_kernel(const float* __restrict__ qg, const float* __restrict__ kg,
                const float* __restrict__ vg, const float* __restrict__ mg,
                float* __restrict__ out, int rowoff, int blend) {
  extern __shared__ char smem[];
  unsigned short* kLDS  = (unsigned short*)smem;          // [256][64] bf16 (key-major)
  unsigned short* vtLDS = kLDS + CS * ND;                 // [64][256] bf16 (V transposed)
  unsigned short* pLDS  = vtLDS + ND * CS;                // 16 waves x [16][32] bf16 staging
  float*          mLDS  = (float*)(pLDS + 16 * 16 * 32);  // [256] mask

  const int chunk = blockIdx.x, hh = blockIdx.y, bb = blockIdx.z;
  const int base_t = rowoff + chunk * CS;
  const int gbase  = (((bb * NH) + hh) * NT + base_t) * ND;

  // ---- cooperative load: K (bf16) and V^T (bf16) via b128 global loads ----
  {
    const float4* kg4 = (const float4*)(kg + gbase);
    const float4* vg4 = (const float4*)(vg + gbase);
#pragma unroll
    for (int i = 0; i < (CS * ND) / (512 * 4); ++i) {     // 8 iters
      int idx4 = threadIdx.x + 512 * i;
      float4 kf = kg4[idx4];
      float4 vf = vg4[idx4];
      int e0 = idx4 * 4;
      int t = e0 >> 6, d = e0 & 63;
      ull kp = (ull)f2bf(kf.x) | ((ull)f2bf(kf.y) << 16) |
               ((ull)f2bf(kf.z) << 32) | ((ull)f2bf(kf.w) << 48);
      *(ull*)(kLDS + e0) = kp;                            // ds_store_b64
      vtLDS[(d + 0) * CS + t] = f2bf(vf.x);
      vtLDS[(d + 1) * CS + t] = f2bf(vf.y);
      vtLDS[(d + 2) * CS + t] = f2bf(vf.z);
      vtLDS[(d + 3) * CS + t] = f2bf(vf.w);
    }
  }
  if (threadIdx.x < CS) mLDS[threadIdx.x] = mg[bb * NT + base_t + threadIdx.x];
  __syncthreads();

  const int wv   = threadIdx.x >> 5;   // wave id: 0..15, one 16-row Q tile each
  const int lane = threadIdx.x & 31;
  const int ln   = lane & 15;
  const int sel  = lane >> 4;
  unsigned short* pst = pLDS + wv * (16 * 32);

  // ---- stage this wave's 16x64 Q tile into two bf16 A-fragments ----
  v16bf qa[2];
#pragma unroll
  for (int h = 0; h < 2; ++h) {
    const float4* q4 = (const float4*)(qg + gbase + (wv * 16) * ND + h * 32);
#pragma unroll
    for (int i = 0; i < 4; ++i) {                         // 16x32 f32 in b128 loads
      int idx4 = i * 32 + lane;
      int row = idx4 >> 3, c4 = (idx4 & 7) * 4;           // q4 row stride = 64/4 = 16
      float4 f = q4[row * 16 + (c4 >> 2)];
      ull qp = (ull)f2bf(f.x) | ((ull)f2bf(f.y) << 16) |
               ((ull)f2bf(f.z) << 32) | ((ull)f2bf(f.w) << 48);
      *(ull*)(pst + row * 32 + c4) = qp;
    }
    qa[h] = load_frag_a(pst, ln, 32, 0, sel);   // same-wave LDS: in-order, no barrier
  }

  // all-ones bf16 B tile (for row-sum-by-WMMA)
  FragU onesf;
#pragma unroll
  for (int e = 0; e < 16; ++e) onesf.u[e] = 0x3F80;

  // ---- online-softmax flash loop over 8 key strips of 32 ----
  v8f o0 = {}, o1 = {}, o2 = {}, o3 = {};       // O tile 16x64 f32 (C layout)
  v8f o4 = {};                                  // l accumulator: P x ones
  float m[8];
#pragma unroll
  for (int r = 0; r < 8; ++r) m[r] = -1e30f;

#pragma unroll 2
  for (int j0 = 0; j0 < CS; j0 += 32) {
    // scores: two 16x16 S tiles, K-contraction over D=64 in two steps
    v8f s0 = {}, s1 = {};
    {
      v16bf kb;
      kb = load_frag_b(kLDS, j0 + ln, ND, 0, sel);
      s0 = __builtin_amdgcn_wmma_f32_16x16x32_bf16(false, qa[0], false, kb, (short)0, s0, false, false);
      kb = load_frag_b(kLDS, j0 + ln, ND, 32, sel);
      s0 = __builtin_amdgcn_wmma_f32_16x16x32_bf16(false, qa[1], false, kb, (short)0, s0, false, false);
      kb = load_frag_b(kLDS, j0 + 16 + ln, ND, 0, sel);
      s1 = __builtin_amdgcn_wmma_f32_16x16x32_bf16(false, qa[0], false, kb, (short)0, s1, false, false);
      kb = load_frag_b(kLDS, j0 + 16 + ln, ND, 32, sel);
      s1 = __builtin_amdgcn_wmma_f32_16x16x32_bf16(false, qa[1], false, kb, (short)0, s1, false, false);
    }

    const float mv0 = mLDS[j0 + ln];
    const float mv1 = mLDS[j0 + 16 + ln];

#pragma unroll
    for (int r = 0; r < 8; ++r) {
      s0[r] = s0[r] * 0.125f + mv0;             // 1/sqrt(64) + mask
      s1[r] = s1[r] * 0.125f + mv1;
      float rm = rowmax16(fmaxf(s0[r], s1[r]));     // DPP rotate-reduce
      float mn   = fmaxf(m[r], rm);
      float corr = __expf(m[r] - mn);
      m[r] = mn;
      float p0 = __expf(s0[r] - mn);
      float p1 = __expf(s1[r] - mn);
      o0[r] *= corr; o1[r] *= corr; o2[r] *= corr; o3[r] *= corr; o4[r] *= corr;
      int prow = r + sel * 8;                   // C-layout row -> staging row
      pst[prow * 32 + ln]      = f2bf(p0);
      pst[prow * 32 + 16 + ln] = f2bf(p1);
    }

    // P (16x32, A layout) x V (32x64): 4 N-tiles, plus P x ones for row-sums
    v16bf pa = load_frag_a(pst, ln, 32, 0, sel);
    v16bf vb;
    vb = load_frag_b(vtLDS,  0 + ln, CS, j0, sel);
    o0 = __builtin_amdgcn_wmma_f32_16x16x32_bf16(false, pa, false, vb, (short)0, o0, false, false);
    vb = load_frag_b(vtLDS, 16 + ln, CS, j0, sel);
    o1 = __builtin_amdgcn_wmma_f32_16x16x32_bf16(false, pa, false, vb, (short)0, o1, false, false);
    vb = load_frag_b(vtLDS, 32 + ln, CS, j0, sel);
    o2 = __builtin_amdgcn_wmma_f32_16x16x32_bf16(false, pa, false, vb, (short)0, o2, false, false);
    vb = load_frag_b(vtLDS, 48 + ln, CS, j0, sel);
    o3 = __builtin_amdgcn_wmma_f32_16x16x32_bf16(false, pa, false, vb, (short)0, o3, false, false);
    o4 = __builtin_amdgcn_wmma_f32_16x16x32_bf16(false, pa, false, onesf.v, (short)0, o4, false, false);
  }

  // ---- finalize and store (optionally blending 0.5*old + 0.5*new) ----
#pragma unroll
  for (int r = 0; r < 8; ++r) {
    float inv = 1.0f / o4[r];                   // l lives in the same C layout as O
    int row   = base_t + wv * 16 + r + sel * 8;
    int obase = (((bb * NH) + hh) * NT + row) * ND;
    float v0 = o0[r] * inv, v1 = o1[r] * inv, v2 = o2[r] * inv, v3 = o3[r] * inv;
    if (blend) {
      out[obase      + ln] = 0.5f * out[obase      + ln] + 0.5f * v0;
      out[obase + 16 + ln] = 0.5f * out[obase + 16 + ln] + 0.5f * v1;
      out[obase + 32 + ln] = 0.5f * out[obase + 32 + ln] + 0.5f * v2;
      out[obase + 48 + ln] = 0.5f * out[obase + 48 + ln] + 0.5f * v3;
    } else {
      out[obase      + ln] = v0;
      out[obase + 16 + ln] = v1;
      out[obase + 32 + ln] = v2;
      out[obase + 48 + ln] = v3;
    }
  }
}

extern "C" void kernel_launch(void* const* d_in, const int* in_sizes, int n_in,
                              void* d_out, int out_size, void* d_ws, size_t ws_size,
                              hipStream_t stream) {
  const float* q    = (const float*)d_in[0];
  const float* k    = (const float*)d_in[1];
  const float* v    = (const float*)d_in[2];
  const float* mask = (const float*)d_in[3];
  float* out = (float*)d_out;

  const size_t smem = (size_t)(CS * ND + ND * CS + 16 * 16 * 32) * sizeof(unsigned short)
                    + (size_t)CS * sizeof(float);   // ~83 KB, well under 320 KB/WGP

  // Raise dynamic-LDS cap (no-op if already sufficient; not a stream op, capture-safe).
  (void)hipFuncSetAttribute((const void*)blk_attn_kernel,
                            hipFuncAttributeMaxDynamicSharedMemorySize, (int)smem);

  // Pass 1: full chunks, overwrite entire output.
  blk_attn_kernel<<<dim3(NT / CS, NH, NB), 512, smem, stream>>>(q, k, v, mask, out, 0, 0);
  // Pass 2: shifted chunks over the middle T-2W rows, blend in place.
  blk_attn_kernel<<<dim3((NT - CS) / CS, NH, NB), 512, smem, stream>>>(q, k, v, mask, out, HW, 1);
}
